// AttentionBlock_420906795141
// MI455X (gfx1250) — compile-verified, matrix operand order checked by reading, not run on hardware
//
#include <hip/hip_runtime.h>
#include <hip/hip_bf16.h>
#include <stdint.h>

#define B_  32
#define C_  256
#define N_  1024
#define G_  8

typedef __attribute__((ext_vector_type(16))) __bf16 v16bf;
typedef __attribute__((ext_vector_type(8)))  float  v8f;
typedef __attribute__((ext_vector_type(4)))  unsigned int v4u;
typedef __attribute__((ext_vector_type(8)))  int    v8i;
typedef __attribute__((ext_vector_type(4)))  int    v4i;

union frag_u { uint4 q[2]; v16bf v; };

__device__ __forceinline__ v8f vzero8() {
  v8f z;
#pragma unroll
  for (int i = 0; i < 8; ++i) z[i] = 0.0f;
  return z;
}

__device__ __forceinline__ uint16_t f2bf(float f) {
  uint32_t u = __float_as_uint(f);
  uint32_t r = u + 0x7FFFu + ((u >> 16) & 1u);   // round-to-nearest-even
  return (uint16_t)(r >> 16);
}

// Load a 16x32 bf16 WMMA fragment from a row-major, K-contiguous matrix
// (global or LDS; address space inferred from the pointer).
// lane l -> row (l&15), K chunk base (l>>4)*8, plus the +16 K half.
__device__ __forceinline__ v16bf load_frag(const uint16_t* p, int ld) {
  int lane = threadIdx.x & 31;
  const uint16_t* q = p + (size_t)(lane & 15) * ld + (lane >> 4) * 8;
  frag_u u;
  u.q[0] = *reinterpret_cast<const uint4*>(q);
  u.q[1] = *reinterpret_cast<const uint4*>(q + 16);
  return u.v;
}

__device__ __forceinline__ v8f wmma_bf16(v16bf a, v16bf b, v8f c) {
  return __builtin_amdgcn_wmma_f32_16x16x32_bf16(false, a, false, b, (short)0, c,
                                                 false, false);
}

// ---- Tensor Data Mover: 2D bf16 tile (tile_d0 x tile_d1, row stride
// stride0 elements) from global -> LDS, packed contiguously at lds_addr.
// D# built per CDNA5 ISA; groups 2/3 zero (dims unused).
// This toolchain exposes the 6-arg builtin: (v4u, v8i, v4i, v4i, v8i, cpol).
__device__ __forceinline__ void tdm_load_2d_bf16(uint32_t lds_addr, const void* gptr,
                                                 uint32_t tile_d0, uint32_t tile_d1,
                                                 uint64_t stride0) {
  uint64_t ga = (uint64_t)(uintptr_t)gptr;
  v4u g0;
  g0[0] = 1u;                                   // count=1, user descriptor
  g0[1] = lds_addr;                             // LDS byte address
  g0[2] = (uint32_t)ga;                         // global_addr[31:0]
  g0[3] = (uint32_t)((ga >> 32) & 0x01FFFFFFu) | (2u << 30);  // addr[56:32], type=2
  const uint32_t td0 = 1u << 22, td1 = 1u << 22;  // huge tensor dims: no OOB clipping
  v8i g1;
  g1[0] = (int)0x00010000u;                     // wg_mask=0, data_size=1 (2 bytes)
  g1[1] = (int)((td0 & 0xFFFFu) << 16);         // atomic_barrier_addr=0, dim0 lo
  g1[2] = (int)(((td0 >> 16) & 0xFFFFu) | ((td1 & 0xFFFFu) << 16));
  g1[3] = (int)(((td1 >> 16) & 0xFFFFu) | (tile_d0 << 16));
  g1[4] = (int)(tile_d1 & 0xFFFFu);             // tile_dim1, tile_dim2=0
  g1[5] = (int)(uint32_t)(stride0 & 0xFFFFFFFFu);          // dim0_stride[31:0]
  g1[6] = (int)(uint32_t)((stride0 >> 32) & 0xFFFFu);      // stride hi, dim1_stride=0
  g1[7] = 0;
  v4i z4;
  z4[0] = 0; z4[1] = 0; z4[2] = 0; z4[3] = 0;
  v8i z8;
#pragma unroll
  for (int i = 0; i < 8; ++i) z8[i] = 0;
  __builtin_amdgcn_tensor_load_to_lds(g0, g1, z4, z4, z8, 0);
}

// ---------------- GroupNorm stats: one block per (b, g) ----------------
__global__ __launch_bounds__(256)
void gn_stats_kernel(const float* __restrict__ x, float* __restrict__ stats) {
  __shared__ float ssum[256];
  __shared__ float ssq[256];
  const int bg  = blockIdx.x;                   // b*8 + g
  const int tid = threadIdx.x;
  const float* base = x + (size_t)bg * 32 * N_; // group region is contiguous
  float s = 0.0f, q = 0.0f;
  for (int i = tid; i < 32 * N_; i += 256) {
    float v = base[i];
    s += v; q += v * v;
  }
  ssum[tid] = s; ssq[tid] = q;
  __syncthreads();
  for (int st = 128; st > 0; st >>= 1) {
    if (tid < st) { ssum[tid] += ssum[tid + st]; ssq[tid] += ssq[tid + st]; }
    __syncthreads();
  }
  if (tid == 0) {
    const float inv = 1.0f / (32.0f * N_);
    float mu  = ssum[0] * inv;
    float var = ssq[0] * inv - mu * mu;
    stats[bg * 2 + 0] = mu;
    stats[bg * 2 + 1] = rsqrtf(var + 1e-5f);
  }
}

// ------------- GroupNorm apply + transpose to hT[b][n][c] (bf16) -------------
__global__ __launch_bounds__(256)
void gn_apply_kernel(const float* __restrict__ x, const float* __restrict__ stats,
                     const float* __restrict__ nw, const float* __restrict__ nb,
                     uint16_t* __restrict__ hT) {
  __shared__ float tile[32][33];
  const int b  = blockIdx.z;
  const int c0 = blockIdx.y * 32;               // aligned to group boundary
  const int n0 = blockIdx.x * 32;
  const int tid = threadIdx.x;
  const int g   = c0 >> 5;
  const float mu = stats[(b * G_ + g) * 2 + 0];
  const float rs = stats[(b * G_ + g) * 2 + 1];
  {
    const int tn  = tid & 31;
    const int tcb = tid >> 5;
#pragma unroll
    for (int i = 0; i < 4; ++i) {
      int c = c0 + tcb + i * 8;
      float v = x[(size_t)b * C_ * N_ + (size_t)c * N_ + n0 + tn];
      tile[tcb + i * 8][tn] = (v - mu) * rs * nw[c] + nb[c];
    }
  }
  __syncthreads();
  {
    const int tc  = tid & 31;
    const int tnb = tid >> 5;
#pragma unroll
    for (int i = 0; i < 4; ++i) {
      int n = n0 + tnb + i * 8;
      hT[(size_t)b * N_ * C_ + (size_t)n * C_ + c0 + tc] = f2bf(tile[tc][tnb + i * 8]);
    }
  }
}

// ---------------- Weight f32 -> bf16 conversion ----------------
__global__ __launch_bounds__(256)
void cvt_w_kernel(const float* __restrict__ qkvw, const float* __restrict__ projw,
                  uint16_t* __restrict__ qkvw_bf, uint16_t* __restrict__ projw_bf) {
  int idx = blockIdx.x * 256 + threadIdx.x;
  if (idx < 768 * 256) qkvw_bf[idx] = f2bf(qkvw[idx]);
  if (idx < 256 * 256) projw_bf[idx] = f2bf(projw[idx]);
}

// ---------------- QKV GEMM: qkv[o,n] = sum_c W[o,c] * h[c,n] ----------------
__global__ __launch_bounds__(256)
void qkv_gemm_kernel(const uint16_t* __restrict__ w, const uint16_t* __restrict__ hT,
                     uint16_t* __restrict__ qT, uint16_t* __restrict__ kT,
                     uint16_t* __restrict__ vM) {
  const int wg   = blockIdx.x * 8 + (threadIdx.x >> 5);
  const int lane = threadIdx.x & 31;
  const int half = lane >> 4;
  const int l15  = lane & 15;
  const int b  = wg / 768;
  const int r  = wg % 768;
  const int ot = r / 16;                        // 0..47
  const int n0 = (r % 16) * 64;
  const uint16_t* hb = hT + (size_t)b * N_ * C_;

  v8f acc[4];
#pragma unroll
  for (int s = 0; s < 4; ++s) acc[s] = vzero8();

#pragma unroll
  for (int kc = 0; kc < 8; ++kc) {
    v16bf af = load_frag(w + (size_t)(ot * 16) * C_ + kc * 32, C_);
#pragma unroll
    for (int s = 0; s < 4; ++s) {
      v16bf bf = load_frag(hb + (size_t)(n0 + s * 16) * C_ + kc * 32, C_);
      acc[s] = wmma_bf16(af, bf, acc[s]);
    }
  }

#pragma unroll
  for (int s = 0; s < 4; ++s) {
    int n = n0 + s * 16 + l15;
#pragma unroll
    for (int g = 0; g < 8; ++g) {
      int o = ot * 16 + g + half * 8;
      uint16_t val = f2bf(acc[s][g]);
      if (o < 256)      qT[(size_t)b * N_ * C_ + (size_t)n * C_ + o]          = val;
      else if (o < 512) kT[(size_t)b * N_ * C_ + (size_t)n * C_ + (o - 256)]  = val;
      else              vM[(size_t)b * C_ * N_ + (size_t)(o - 512) * N_ + n]  = val;
    }
  }
}

// ---------------- Flash attention with TDM-staged K/V ----------------
// Block = 8 waves, 128 q-rows of one batch. Wave 0 drives the Tensor Data
// Mover to double-buffer the 32x256 K tile and 256x32 V tile into LDS; the
// DMA for j-pair jp+1 overlaps the 32 WMMAs of j-pair jp.
__global__ __launch_bounds__(256)
void flash_attn_kernel(const uint16_t* __restrict__ qT, const uint16_t* __restrict__ kT,
                       const uint16_t* __restrict__ vM, uint16_t* __restrict__ ao) {
  __shared__ uint16_t sh_k[2][32 * 256];        // 2 x 16 KB, rows of kT[b]
  __shared__ uint16_t sh_v[2][256 * 32];        // 2 x 16 KB, [c][j] packed ld=32
  __shared__ uint16_t plds[8][16 * 32];         // per-wave P staging
  const int wave = threadIdx.x >> 5;
  const int lane = threadIdx.x & 31;
  const int half = lane >> 4;
  const int l15  = lane & 15;
  const int b    = blockIdx.x >> 3;
  const int rg   = blockIdx.x & 7;
  const int row0 = rg * 128 + wave * 16;

  const uint16_t* qb = qT + (size_t)b * N_ * C_ + (size_t)row0 * C_;
  const uint16_t* kb = kT + (size_t)b * N_ * C_;
  const uint16_t* vb = vM + (size_t)b * C_ * N_;

  v16bf qf[8];
#pragma unroll
  for (int kc = 0; kc < 8; ++kc) qf[kc] = load_frag(qb + kc * 32, C_);

  v8f oacc[16];
#pragma unroll
  for (int t = 0; t < 16; ++t) oacc[t] = vzero8();
  float mrun[8], lrun[8];
#pragma unroll
  for (int g = 0; g < 8; ++g) { mrun[g] = -1e30f; lrun[g] = 0.0f; }

  const float scale = 0.0625f;                  // C^-0.5 = 1/16

  // Prologue: DMA j-pair 0 into buffer 0 (wave 0 only; EXEC-independent op,
  // TENSORcnt is per-wave so the issuing wave also waits).
  if (threadIdx.x < 32) {
    tdm_load_2d_bf16((uint32_t)(uintptr_t)&sh_k[0][0], kb,        256, 32, C_);
    tdm_load_2d_bf16((uint32_t)(uintptr_t)&sh_v[0][0], vb,        32, 256, N_);
  }

  for (int jp = 0; jp < 32; ++jp) {
    const int buf = jp & 1;
    if (threadIdx.x < 32) __builtin_amdgcn_s_wait_tensorcnt(0);
    __syncthreads();                            // tiles for jp are visible to all
    if (jp + 1 < 32 && threadIdx.x < 32) {      // prefetch next j-pair
      const uint16_t* kn = kb + (size_t)((jp + 1) * 32) * C_;
      const uint16_t* vn = vb + (size_t)(jp + 1) * 32;
      tdm_load_2d_bf16((uint32_t)(uintptr_t)&sh_k[buf ^ 1][0], kn, 256, 32, C_);
      tdm_load_2d_bf16((uint32_t)(uintptr_t)&sh_v[buf ^ 1][0], vn, 32, 256, N_);
    }

    v8f s0 = vzero8(), s1 = vzero8();
#pragma unroll
    for (int kc = 0; kc < 8; ++kc)
      s0 = wmma_bf16(qf[kc], load_frag(&sh_k[buf][kc * 32], 256), s0);
#pragma unroll
    for (int kc = 0; kc < 8; ++kc)
      s1 = wmma_bf16(qf[kc], load_frag(&sh_k[buf][16 * 256 + kc * 32], 256), s1);

    // Online softmax. D-layout: comp g, lanes 0-15 -> row g; lanes 16-31 -> row g+8.
    // xor masks < 16 keep the two 16-lane halves independent -> per-row reductions.
#pragma unroll
    for (int g = 0; g < 8; ++g) {
      float a  = s0[g] * scale;
      float bb = s1[g] * scale;
      float tm = fmaxf(a, bb);
      tm = fmaxf(tm, __shfl_xor(tm, 1));
      tm = fmaxf(tm, __shfl_xor(tm, 2));
      tm = fmaxf(tm, __shfl_xor(tm, 4));
      tm = fmaxf(tm, __shfl_xor(tm, 8));
      float nm    = fmaxf(mrun[g], tm);
      float alpha = __expf(mrun[g] - nm);
      mrun[g] = nm;
      float p0 = __expf(a - nm);
      float p1 = __expf(bb - nm);
      float ls = p0 + p1;
      ls += __shfl_xor(ls, 1);
      ls += __shfl_xor(ls, 2);
      ls += __shfl_xor(ls, 4);
      ls += __shfl_xor(ls, 8);
      lrun[g] = lrun[g] * alpha + ls;
#pragma unroll
      for (int t = 0; t < 16; ++t) oacc[t][g] *= alpha;
      int row = g + half * 8;
      plds[wave][row * 32 + l15]      = f2bf(p0);
      plds[wave][row * 32 + 16 + l15] = f2bf(p1);
    }
    asm volatile("s_wait_dscnt 0x0" ::: "memory");

    // Re-read P as an A-fragment (C-layout -> A-layout via LDS).
    frag_u pu;
    const uint16_t* pp = &plds[wave][l15 * 32 + half * 8];
    pu.q[0] = *reinterpret_cast<const uint4*>(pp);
    pu.q[1] = *reinterpret_cast<const uint4*>(pp + 16);
    v16bf pf = pu.v;

#pragma unroll
    for (int t = 0; t < 16; ++t) {
      v16bf vf = load_frag(&sh_v[buf][t * 16 * 32], 32);
      oacc[t] = wmma_bf16(pf, vf, oacc[t]);
    }
    __syncthreads();                            // all done with buf before reuse
  }

  uint16_t* aob = ao + (size_t)b * N_ * C_ + (size_t)row0 * C_;
#pragma unroll
  for (int g = 0; g < 8; ++g) {
    float inv = 1.0f / lrun[g];
    int row = g + half * 8;
#pragma unroll
    for (int t = 0; t < 16; ++t)
      aob[(size_t)row * C_ + t * 16 + l15] = f2bf(oacc[t][g] * inv);
  }
}

// ---------------- Proj GEMM + bias + residual ----------------
__global__ __launch_bounds__(256)
void proj_gemm_kernel(const uint16_t* __restrict__ w, const uint16_t* __restrict__ ao,
                      const float* __restrict__ pb, const float* __restrict__ x,
                      float* __restrict__ out) {
  const int wg   = blockIdx.x * 8 + (threadIdx.x >> 5);
  const int lane = threadIdx.x & 31;
  const int half = lane >> 4;
  const int l15  = lane & 15;
  const int b  = wg / 256;
  const int r  = wg % 256;
  const int ot = r / 16;                        // 0..15
  const int n0 = (r % 16) * 64;
  const uint16_t* ab = ao + (size_t)b * N_ * C_;

  v8f acc[4];
#pragma unroll
  for (int s = 0; s < 4; ++s) acc[s] = vzero8();

#pragma unroll
  for (int kc = 0; kc < 8; ++kc) {
    v16bf af = load_frag(w + (size_t)(ot * 16) * C_ + kc * 32, C_);
#pragma unroll
    for (int s = 0; s < 4; ++s) {
      v16bf bf = load_frag(ab + (size_t)(n0 + s * 16) * C_ + kc * 32, C_);
      acc[s] = wmma_bf16(af, bf, acc[s]);
    }
  }

#pragma unroll
  for (int s = 0; s < 4; ++s) {
    int n = n0 + s * 16 + l15;
#pragma unroll
    for (int g = 0; g < 8; ++g) {
      int o = ot * 16 + g + half * 8;
      size_t idx = (size_t)b * C_ * N_ + (size_t)o * N_ + n;
      out[idx] = x[idx] + pb[o] + acc[s][g];
    }
  }
}

extern "C" void kernel_launch(void* const* d_in, const int* in_sizes, int n_in,
                              void* d_out, int out_size, void* d_ws, size_t ws_size,
                              hipStream_t stream) {
  (void)in_sizes; (void)n_in; (void)out_size; (void)ws_size;
  const float* x     = (const float*)d_in[0];
  const float* nw    = (const float*)d_in[1];
  const float* nb    = (const float*)d_in[2];
  const float* qkvw  = (const float*)d_in[3];
  const float* projw = (const float*)d_in[4];
  const float* pb    = (const float*)d_in[5];
  float* out = (float*)d_out;

  char* ws = (char*)d_ws;
  float*    stats    = (float*)   (ws + 0);                        //   2 KB
  uint16_t* qkvw_bf  = (uint16_t*)(ws + 4096);                     // 384 KB
  uint16_t* projw_bf = (uint16_t*)(ws + 4096 + 768 * 256 * 2);     // 128 KB
  uint16_t* hT = (uint16_t*)(ws + (size_t)1  * 1024 * 1024);       // 16 MB, [B,N,C] bf16
  uint16_t* qT = (uint16_t*)(ws + (size_t)17 * 1024 * 1024);       // 16 MB, [B,N,C]
  uint16_t* kT = (uint16_t*)(ws + (size_t)33 * 1024 * 1024);       // 16 MB, [B,N,C]
  uint16_t* vM = (uint16_t*)(ws + (size_t)49 * 1024 * 1024);       // 16 MB, [B,C,N]
  uint16_t* ao = hT;                                               // hT dead after QKV

  gn_stats_kernel<<<B_ * G_, 256, 0, stream>>>(x, stats);
  cvt_w_kernel<<<768, 256, 0, stream>>>(qkvw, projw, qkvw_bf, projw_bf);
  gn_apply_kernel<<<dim3(N_ / 32, C_ / 32, B_), 256, 0, stream>>>(x, stats, nw, nb, hT);
  qkv_gemm_kernel<<<(B_ * 48 * 16) / 8, 256, 0, stream>>>(qkvw_bf, hT, qT, kT, vM);
  flash_attn_kernel<<<B_ * (N_ / 128), 256, 0, stream>>>(qT, kT, vM, ao);
  proj_gemm_kernel<<<(B_ * 16 * 16) / 8, 256, 0, stream>>>(projw_bf, ao, pb, x, out);
}
